// RoPEAttention_6880537608619
// MI455X (gfx1250) — matrix operand order repaired and checked by reading, not compiled, not run on hardware
//
#include <hip/hip_runtime.h>
#include <hip/hip_bf16.h>

typedef __attribute__((ext_vector_type(16))) __bf16 v16bf;
typedef __attribute__((ext_vector_type(8)))  __bf16 v8bf;
typedef __attribute__((ext_vector_type(8)))  float  v8f;

#define EMBED 1024
#define NHEAD 16
#define HDIM  64
#define BATCH 4
#define SEQ   2048
#define MROWS (BATCH*SEQ)   // 8192

// ---------------------------------------------------------------- WMMA helper
__device__ __forceinline__ v8f wmma_bf16(v16bf a, v16bf b, v8f c) {
  // (neg_a, A, neg_b, B, c_mod, C, reuse_a, reuse_b)
  return __builtin_amdgcn_wmma_f32_16x16x32_bf16(false, a, false, b, (short)0, c,
                                                 false, false);
}

// A-matrix fragment (16x32, M x K), contiguous k order used here (not ISA table order):
// K = i + 16*(lane>=16) (row = lane%16), matching B frags
__device__ __forceinline__ v16bf load_a_frag(const __bf16* row, int koff, int sel) {
  return *reinterpret_cast<const v16bf*>(row + koff + 16 * sel);
}

// B-matrix fragment (32x16, K x N), contiguous k order used here (not ISA table order):
// lane = column N (mod 16), K = i + 16*(lane>=16) -> one contiguous 32B load.
__device__ __forceinline__ v16bf load_b_frag(const __bf16* row, int koff, int sel) {
  return *reinterpret_cast<const v16bf*>(row + koff + 16 * sel);
}

// ---------------------------------------------------- async B-tile staging
// Copy 128 rows x 32 cols (bf16) of W (rows nblk..nblk+127, cols k0..k0+31)
// into LDS using the CDNA5 async-to-LDS engine (tracked by ASYNCcnt).
// 512 x 16B chunks, 256 threads -> 2 async b128 issues per thread.
__device__ __forceinline__ void stage_b_tile(const __bf16* __restrict__ W,
                                             int nblk, int k0,
                                             __bf16* dstBase, int tid) {
#pragma unroll
  for (int it = 0; it < 2; ++it) {
    int c   = tid + it * 256;       // 0..511
    int row = c >> 2;
    int qw  = (c & 3) * 8;          // element offset within row (16B chunks)
    const __bf16* g = W + (size_t)(nblk + row) * EMBED + k0 + qw;
    unsigned           laddr = (unsigned)(uintptr_t)(dstBase + row * 32 + qw);
    unsigned long long gaddr = (unsigned long long)(uintptr_t)g;
    asm volatile("global_load_async_to_lds_b128 %0, %1, off"
                 :: "v"(laddr), "v"(gaddr) : "memory");
  }
}

// ------------------------------------------------------------ GEMM core loop
// Wave tile 32(M) x 64(N); block tile 128x128; B from double-buffered LDS.
__device__ __forceinline__ void gemm_core(const __bf16* __restrict__ X,
                                          const __bf16* __restrict__ W,
                                          __bf16* ldsB,  // [2][128][32]
                                          int m0, int wn, int nblk,
                                          int tid, int sel, int r16,
                                          v8f c[2][4]) {
#pragma unroll
  for (int i = 0; i < 2; ++i)
#pragma unroll
    for (int j = 0; j < 4; ++j)
#pragma unroll
      for (int v = 0; v < 8; ++v) c[i][j][v] = 0.f;

  const __bf16* arow0 = X + (size_t)(m0 + r16) * EMBED;
  const __bf16* arow1 = arow0 + (size_t)16 * EMBED;

  stage_b_tile(W, nblk, 0, ldsB, tid);

  for (int k0 = 0; k0 < EMBED; k0 += 32) {
    const int buf = (k0 >> 5) & 1;
    if (k0 + 32 < EMBED) {
      stage_b_tile(W, nblk, k0 + 32, ldsB + (buf ^ 1) * 128 * 32, tid);
      asm volatile("s_wait_asynccnt 2" ::: "memory");  // previous slice done
    } else {
      asm volatile("s_wait_asynccnt 0" ::: "memory");
    }
    __syncthreads();  // B tile visible to all waves

    __builtin_prefetch(arow0 + k0 + 128, 0, 0);
    __builtin_prefetch(arow1 + k0 + 128, 0, 0);
    v16bf a0 = load_a_frag(arow0, k0, sel);
    v16bf a1 = load_a_frag(arow1, k0, sel);
    const __bf16* bbase = ldsB + buf * 128 * 32;
#pragma unroll
    for (int ni = 0; ni < 4; ++ni) {
      v16bf bf = load_b_frag(bbase + (size_t)(wn * 64 + ni * 16 + r16) * 32, 0, sel);
      c[0][ni] = wmma_bf16(a0, bf, c[0][ni]);
      c[1][ni] = wmma_bf16(a1, bf, c[1][ni]);
    }
    __syncthreads();  // all reads done before buffer reuse
  }
}

// ---------------------------------------------------------------- cast kernel
__global__ void cast_f32_bf16(const float* __restrict__ src,
                              __bf16* __restrict__ dst, int n) {
  int i = blockIdx.x * blockDim.x + threadIdx.x;
  if (i < n) dst[i] = (__bf16)src[i];
}

// ------------------------------------------------- QKV projection + RoPE/store
// Y = X @ W^T + bias.  rope_mode=1: apply RoPE, store [B,H,L,64].
// rope_mode=0 (V): store transposed [B,H,64,L].
__global__ __launch_bounds__(256) void qkv_gemm_rope(
    const __bf16* __restrict__ X, const __bf16* __restrict__ W,
    const float* __restrict__ bias, __bf16* __restrict__ Out, int rope_mode) {
  __shared__ __bf16 ldsB[2 * 128 * 32];  // 16 KB double-buffered B tile
  const int tid  = threadIdx.x;
  const int lane = tid & 31;
  const int w    = tid >> 5;
  const int sel  = lane >> 4;
  const int r16  = lane & 15;
  const int wm = w >> 1, wn = w & 1;
  const int m0   = blockIdx.y * 128 + wm * 32;
  const int nblk = blockIdx.x * 128;
  const int n0   = nblk + wn * 64;

  v8f c[2][4];
  gemm_core(X, W, ldsB, m0, wn, nblk, tid, sel, r16, c);

  const float L2B = 13.287712379549449f / 64.0f;  // log2(10000)/HDIM
#pragma unroll
  for (int mi = 0; mi < 2; ++mi)
#pragma unroll
    for (int ni = 0; ni < 4; ++ni) {
      const int n = n0 + ni * 16 + r16;
      const float bv = bias[n];
      const int h = n >> 6, d = n & 63;
      const float sgn  = (d & 1) ? 1.f : -1.f;
      const float freq = __builtin_exp2f(-(float)(d & ~1) * L2B);
#pragma unroll
      for (int v = 0; v < 8; ++v) {
        const int m = m0 + mi * 16 + sel * 8 + v;
        const int b = m >> 11, l = m & 2047;
        float val = c[mi][ni][v] + bv;
        if (rope_mode) {
          float partner = __shfl_xor(val, 1, 32);  // RoPE pair lives next lane
          float ang = (float)l * freq;
          float sn = __builtin_sinf(ang);
          float cs = __builtin_cosf(ang);
          float res = val * cs + sgn * partner * sn;
          Out[(((size_t)(b * NHEAD + h) * SEQ + l) << 6) + d] = (__bf16)res;
        } else {  // V: transposed store for contiguous PV B-frags
          Out[((size_t)(b * NHEAD + h) * HDIM + d) * SEQ + l] = (__bf16)val;
        }
      }
    }
}

// --------------------------------------------------- output projection (fp32)
__global__ __launch_bounds__(256) void proj_gemm(
    const __bf16* __restrict__ X, const __bf16* __restrict__ W,
    const float* __restrict__ bias, float* __restrict__ Out) {
  __shared__ __bf16 ldsB[2 * 128 * 32];
  const int tid  = threadIdx.x;
  const int lane = tid & 31;
  const int w    = tid >> 5;
  const int sel  = lane >> 4;
  const int r16  = lane & 15;
  const int wm = w >> 1, wn = w & 1;
  const int m0   = blockIdx.y * 128 + wm * 32;
  const int nblk = blockIdx.x * 128;
  const int n0   = nblk + wn * 64;

  v8f c[2][4];
  gemm_core(X, W, ldsB, m0, wn, nblk, tid, sel, r16, c);

#pragma unroll
  for (int mi = 0; mi < 2; ++mi)
#pragma unroll
    for (int ni = 0; ni < 4; ++ni) {
      const int n = n0 + ni * 16 + r16;
      const float bv = bias[n];
#pragma unroll
      for (int v = 0; v < 8; ++v) {
        const int m = m0 + mi * 16 + sel * 8 + v;
        Out[(size_t)m * EMBED + n] = c[mi][ni][v] + bv;
      }
    }
}

// ------------------------------------------------------- flash attention core
__global__ __launch_bounds__(256) void flash_attn(
    const __bf16* __restrict__ Q, const __bf16* __restrict__ K,
    const __bf16* __restrict__ Vt, __bf16* __restrict__ AO) {
  __shared__ __bf16 lds[8 * 32 * 32];  // per-wave 32x32 P staging (16 KB)
  const int lane = threadIdx.x & 31;
  const int w    = threadIdx.x >> 5;
  const int sel  = lane >> 4;
  const int r16  = lane & 15;
  const int b = blockIdx.z, h = blockIdx.y;
  const int m0 = blockIdx.x * 256 + w * 32;  // query row within L

  const __bf16* Qh = Q  + (size_t)(b * NHEAD + h) * SEQ * HDIM;
  const __bf16* Kh = K  + (size_t)(b * NHEAD + h) * SEQ * HDIM;
  const __bf16* Vh = Vt + (size_t)(b * NHEAD + h) * HDIM * SEQ;
  __bf16* ldsW = lds + w * 32 * 32;

  // Q A-frags resident across whole key loop
  v16bf qa[2][2];
#pragma unroll
  for (int mi = 0; mi < 2; ++mi)
#pragma unroll
    for (int kk = 0; kk < 2; ++kk)
      qa[mi][kk] = load_a_frag(Qh + (size_t)(m0 + mi * 16 + r16) * HDIM, kk * 32, sel);

  float mst[2][8], lst[2][8];
  v8f o[2][4];
#pragma unroll
  for (int mi = 0; mi < 2; ++mi)
#pragma unroll
    for (int v = 0; v < 8; ++v) { mst[mi][v] = -1e30f; lst[mi][v] = 0.f; }
#pragma unroll
  for (int mi = 0; mi < 2; ++mi)
#pragma unroll
    for (int nj = 0; nj < 4; ++nj)
#pragma unroll
      for (int v = 0; v < 8; ++v) o[mi][nj][v] = 0.f;

  for (int j0 = 0; j0 <= m0; j0 += 32) {
    // K B-frags: S = Q @ K^T, so B[k][n] = K[n][k] -> contiguous K-row loads
    v16bf kb[2][2];
#pragma unroll
    for (int ni = 0; ni < 2; ++ni)
#pragma unroll
      for (int kk = 0; kk < 2; ++kk)
        kb[ni][kk] = load_b_frag(Kh + (size_t)(j0 + ni * 16 + r16) * HDIM, kk * 32, sel);

    v8f s[2][2];
#pragma unroll
    for (int mi = 0; mi < 2; ++mi)
#pragma unroll
      for (int ni = 0; ni < 2; ++ni) {
        v8f acc;
#pragma unroll
        for (int v = 0; v < 8; ++v) acc[v] = 0.f;
        acc = wmma_bf16(qa[mi][0], kb[ni][0], acc);
        acc = wmma_bf16(qa[mi][1], kb[ni][1], acc);
        s[mi][ni] = acc;
      }

    const bool diag = (j0 == m0);
#pragma unroll
    for (int mi = 0; mi < 2; ++mi)
#pragma unroll
      for (int ni = 0; ni < 2; ++ni)
#pragma unroll
        for (int v = 0; v < 8; ++v) {
          float sc = s[mi][ni][v] * 0.125f;  // 1/sqrt(64)
          if (diag && (ni * 16 + r16) > (mi * 16 + sel * 8 + v)) sc = -1e30f;
          s[mi][ni][v] = sc;
        }

    // online softmax (row = (v, lane-half); reduce over 16-lane half groups)
#pragma unroll
    for (int mi = 0; mi < 2; ++mi) {
      float rmax[8], rsum[8], al[8];
#pragma unroll
      for (int v = 0; v < 8; ++v) rmax[v] = __builtin_fmaxf(s[mi][0][v], s[mi][1][v]);
#pragma unroll
      for (int off = 1; off <= 8; off <<= 1)
#pragma unroll
        for (int v = 0; v < 8; ++v)
          rmax[v] = __builtin_fmaxf(rmax[v], __shfl_xor(rmax[v], off, 32));
#pragma unroll
      for (int v = 0; v < 8; ++v) {
        float nm = __builtin_fmaxf(mst[mi][v], rmax[v]);
        al[v] = __builtin_expf(mst[mi][v] - nm);
        mst[mi][v] = nm;
        float p0 = __builtin_expf(s[mi][0][v] - nm);
        float p1 = __builtin_expf(s[mi][1][v] - nm);
        s[mi][0][v] = p0; s[mi][1][v] = p1;
        rsum[v] = p0 + p1;
      }
#pragma unroll
      for (int off = 1; off <= 8; off <<= 1)
#pragma unroll
        for (int v = 0; v < 8; ++v) rsum[v] += __shfl_xor(rsum[v], off, 32);
#pragma unroll
      for (int v = 0; v < 8; ++v) lst[mi][v] = lst[mi][v] * al[v] + rsum[v];
#pragma unroll
      for (int nj = 0; nj < 4; ++nj)
#pragma unroll
        for (int v = 0; v < 8; ++v) o[mi][nj][v] *= al[v];
    }

    // C-layout -> A-layout transpose of P through private LDS
#pragma unroll
    for (int mi = 0; mi < 2; ++mi)
#pragma unroll
      for (int ni = 0; ni < 2; ++ni)
#pragma unroll
        for (int v = 0; v < 8; ++v)
          ldsW[(mi * 16 + sel * 8 + v) * 32 + ni * 16 + r16] = (__bf16)s[mi][ni][v];
    asm volatile("s_wait_dscnt 0" ::: "memory");
    v16bf pa0 = load_a_frag(ldsW + (size_t)(r16) * 32, 0, sel);
    v16bf pa1 = load_a_frag(ldsW + (size_t)(16 + r16) * 32, 0, sel);

    // PV: B[k][n] = V[j0+k][n] = Vt[n][j0+k] -> contiguous Vt-row loads
#pragma unroll
    for (int nj = 0; nj < 4; ++nj) {
      v16bf vb = *reinterpret_cast<const v16bf*>(
          Vh + (size_t)(nj * 16 + r16) * SEQ + j0 + 16 * sel);
      o[0][nj] = wmma_bf16(pa0, vb, o[0][nj]);
      o[1][nj] = wmma_bf16(pa1, vb, o[1][nj]);
    }
  }

  // finalize: divide by l, store head-merged [B, L, E] bf16
#pragma unroll
  for (int mi = 0; mi < 2; ++mi)
#pragma unroll
    for (int nj = 0; nj < 4; ++nj)
#pragma unroll
      for (int v = 0; v < 8; ++v) {
        int q = m0 + mi * 16 + sel * 8 + v;
        int d = nj * 16 + r16;
        float val = o[mi][nj][v] / lst[mi][v];
        AO[((size_t)(b * SEQ + q)) * EMBED + h * HDIM + d] = (__bf16)val;
      }
}

// ----------------------------------------------------------------- launcher
extern "C" void kernel_launch(void* const* d_in, const int* in_sizes, int n_in,
                              void* d_out, int out_size, void* d_ws, size_t ws_size,
                              hipStream_t stream) {
  const float* x  = (const float*)d_in[0];
  // d_in[1] = attn_mask (causal triu) -> handled analytically, not read
  const float* Wq = (const float*)d_in[2];
  const float* bq = (const float*)d_in[3];
  const float* Wk = (const float*)d_in[4];
  const float* bk = (const float*)d_in[5];
  const float* Wv = (const float*)d_in[6];
  const float* bv = (const float*)d_in[7];
  const float* Wp = (const float*)d_in[8];
  const float* bp = (const float*)d_in[9];
  float* out = (float*)d_out;

  char* ws = (char*)d_ws;
  size_t off = 0;
  __bf16* xb  = (__bf16*)(ws + off); off += (size_t)MROWS * EMBED * 2;
  __bf16* wqb = (__bf16*)(ws + off); off += (size_t)EMBED * EMBED * 2;
  __bf16* wkb = (__bf16*)(ws + off); off += (size_t)EMBED * EMBED * 2;
  __bf16* wvb = (__bf16*)(ws + off); off += (size_t)EMBED * EMBED * 2;
  __bf16* wpb = (__bf16*)(ws + off); off += (size_t)EMBED * EMBED * 2;
  __bf16* Qb  = (__bf16*)(ws + off); off += (size_t)MROWS * EMBED * 2;
  __bf16* Kb  = (__bf16*)(ws + off); off += (size_t)MROWS * EMBED * 2;
  __bf16* Vtb = (__bf16*)(ws + off); off += (size_t)MROWS * EMBED * 2;
  __bf16* AOb = (__bf16*)(ws + off); off += (size_t)MROWS * EMBED * 2;
  (void)ws_size; (void)in_sizes; (void)n_in; (void)out_size;

  const int NX = MROWS * EMBED;
  const int NW = EMBED * EMBED;
  cast_f32_bf16<<<(NX + 255) / 256, 256, 0, stream>>>(x,  xb,  NX);
  cast_f32_bf16<<<(NW + 255) / 256, 256, 0, stream>>>(Wq, wqb, NW);
  cast_f32_bf16<<<(NW + 255) / 256, 256, 0, stream>>>(Wk, wkb, NW);
  cast_f32_bf16<<<(NW + 255) / 256, 256, 0, stream>>>(Wv, wvb, NW);
  cast_f32_bf16<<<(NW + 255) / 256, 256, 0, stream>>>(Wp, wpb, NW);

  dim3 ggrid(EMBED / 128, MROWS / 128);  // (8, 64)
  qkv_gemm_rope<<<ggrid, 256, 0, stream>>>(xb, wqb, bq, Qb, 1);
  qkv_gemm_rope<<<ggrid, 256, 0, stream>>>(xb, wkb, bk, Kb, 1);
  qkv_gemm_rope<<<ggrid, 256, 0, stream>>>(xb, wvb, bv, Vtb, 0);

  flash_attn<<<dim3(SEQ / 256, NHEAD, BATCH), 256, 0, stream>>>(Qb, Kb, Vtb, AOb);

  proj_gemm<<<ggrid, 256, 0, stream>>>(AOb, wpb, bp, out);
}
